// CrossAttention_82291573391928
// MI455X (gfx1250) — matrix operand order repaired and checked by reading, not compiled, not run on hardware
//
#include <hip/hip_runtime.h>

typedef _Float16 half_t;
typedef __attribute__((ext_vector_type(16))) _Float16 v16h;
typedef __attribute__((ext_vector_type(8)))  _Float16 v8h;
typedef __attribute__((ext_vector_type(4)))  _Float16 v4h;
typedef __attribute__((ext_vector_type(8)))  float    v8f;

#define WMMA_F32_F16(a, b, c) \
  __builtin_amdgcn_wmma_f32_16x16x32_f16(false, (a), false, (b), (short)0, (c), false, false)

static __device__ __forceinline__ v16h cat8(v8h lo, v8h hi) {
  return __builtin_shufflevector(lo, hi, 0,1,2,3,4,5,6,7,8,9,10,11,12,13,14,15);
}

// ---- CDNA5 async global->LDS copy (ASYNCcnt tracked), per-lane 16B ----------
static __device__ __forceinline__ void async_b128(unsigned lds, const half_t* g) {
  asm volatile("global_load_async_to_lds_b128 %0, %1, off"
               :: "v"(lds), "v"((unsigned long long)(uintptr_t)g) : "memory");
}
static __device__ __forceinline__ void wait_async0() {
  asm volatile("s_wait_asynccnt 0x0" ::: "memory");
}
static __device__ __forceinline__ unsigned lds_off(const half_t* p) {
  return (unsigned)(uintptr_t)p;  // low 32 bits of generic ptr = LDS address
}

// ---- DPP16 row (16-lane) all-reduce: quad xor1, xor2, then ror4, ror8 -------
template <int CTRL>
static __device__ __forceinline__ float dpp_mov(float x) {
  return __int_as_float(
      __builtin_amdgcn_update_dpp(0, __float_as_int(x), CTRL, 0xF, 0xF, true));
}
static __device__ __forceinline__ float row_max16(float x) {
  x = fmaxf(x, dpp_mov<0xB1>(x));   // quad_perm [1,0,3,2]
  x = fmaxf(x, dpp_mov<0x4E>(x));   // quad_perm [2,3,0,1]
  x = fmaxf(x, dpp_mov<0x124>(x));  // row_ror:4
  x = fmaxf(x, dpp_mov<0x128>(x));  // row_ror:8
  return x;
}
static __device__ __forceinline__ float row_sum16(float x) {
  x += dpp_mov<0xB1>(x);
  x += dpp_mov<0x4E>(x);
  x += dpp_mov<0x124>(x);
  x += dpp_mov<0x128>(x);
  return x;
}

// ---------------------------------------------------------------------------
// f32 -> f16 elementwise convert (4 elems / thread)
// ---------------------------------------------------------------------------
__global__ void cvt_f32_to_f16(const float* __restrict__ in,
                               half_t* __restrict__ out, int n4) {
  int i = blockIdx.x * blockDim.x + threadIdx.x;
  if (i < n4) {
    float4 v = ((const float4*)in)[i];
    v4h h;
    h[0] = (half_t)v.x; h[1] = (half_t)v.y; h[2] = (half_t)v.z; h[3] = (half_t)v.w;
    ((v4h*)out)[i] = h;
  }
}

// ---------------------------------------------------------------------------
// f32 (K x N) -> f16 transposed (N x K) with optional scale folding.
// ---------------------------------------------------------------------------
__global__ void cvt_transpose_f16(const float* __restrict__ in,
                                  half_t* __restrict__ out, int K, int N,
                                  float scale) {
  int n = blockIdx.x * blockDim.x + threadIdx.x;
  int k = blockIdx.y;
  if (n < N) out[(size_t)n * K + k] = (half_t)(in[(size_t)k * N + n] * scale);
}

// ---------------------------------------------------------------------------
// WMMA GEMM, double-buffered async staging.
// C[M x N] = A[M x K] * Bt[N x K]^T.  Block tile 128x128, 8 waves (2x4),
// each wave 64x32 via 4x2 WMMA 16x16x32.  OUT16: f16 out, else f32+bias.
// ---------------------------------------------------------------------------
template <int OUT16>
__global__ __launch_bounds__(256)
void gemm_wmma(const half_t* __restrict__ A, const half_t* __restrict__ Bt,
               half_t* __restrict__ C16, float* __restrict__ C32,
               const float* __restrict__ bias, int Mdim, int Ndim, int Kdim) {
  constexpr int AP = 40;  // 32 + 8 pad halves (80B pitch, 16B aligned)
  __shared__ __align__(16) half_t As[2][128 * AP];
  __shared__ __align__(16) half_t Bs[2][128 * AP];

  const int tid  = threadIdx.x;
  const int lane = tid & 31;
  const int w    = tid >> 5;
  const int wm   = w >> 2;
  const int wn   = w & 3;
  const int m0   = blockIdx.x * 128;
  const int n0   = blockIdx.y * 128;

  v8f acc[4][2];
  v8f zero = {};
#pragma unroll
  for (int i = 0; i < 4; i++)
#pragma unroll
    for (int j = 0; j < 2; j++) acc[i][j] = zero;

  // async staging: tile = 128 rows x 32 halves = 512 x b128; 2 per thread
  auto stage = [&](int buf, int k0) {
#pragma unroll
    for (int i = 0; i < 2; i++) {
      int j = tid * 2 + i;
      int row = j >> 2, s = (j & 3) * 8;
      async_b128(lds_off(&As[buf][row * AP + s]),
                 A + (size_t)(m0 + row) * Kdim + k0 + s);
      async_b128(lds_off(&Bs[buf][row * AP + s]),
                 Bt + (size_t)(n0 + row) * Kdim + k0 + s);
    }
  };

  stage(0, 0);
  const int ksel = (lane < 16) ? 0 : 8;
  const int kgrp = (lane >> 4) * 8;

  int buf = 0;
  for (int k0 = 0; k0 < Kdim; k0 += 32, buf ^= 1) {
    wait_async0();
    __syncthreads();
    if (k0 + 32 < Kdim) stage(buf ^ 1, k0 + 32);

    v16h afr[4];
#pragma unroll
    for (int mf = 0; mf < 4; mf++) {
      int r = wm * 64 + mf * 16 + (lane & 15);
      v8h lo = *((const v8h*)&As[buf][r * AP + ksel]);
      v8h hi = *((const v8h*)&As[buf][r * AP + ksel + 16]);
      afr[mf] = cat8(lo, hi);
    }
#pragma unroll
    for (int nf = 0; nf < 2; nf++) {
      int r = wn * 32 + nf * 16 + (lane & 15);
      v8h lo = *((const v8h*)&Bs[buf][r * AP + kgrp]);
      v8h hi = *((const v8h*)&Bs[buf][r * AP + kgrp + 16]);
      v16h bfr = cat8(lo, hi);
#pragma unroll
      for (int mf = 0; mf < 4; mf++)
        acc[mf][nf] = WMMA_F32_F16(afr[mf], bfr, acc[mf][nf]);
    }
  }

  const int ch = lane >> 4;
  const int cn = lane & 15;
#pragma unroll
  for (int mf = 0; mf < 4; mf++)
#pragma unroll
    for (int nf = 0; nf < 2; nf++)
#pragma unroll
      for (int v = 0; v < 8; v++) {
        int m = m0 + wm * 64 + mf * 16 + v + ch * 8;
        int n = n0 + wn * 32 + nf * 16 + cn;
        if (OUT16) C16[(size_t)m * Ndim + n] = (half_t)acc[mf][nf][v];
        else       C32[(size_t)m * Ndim + n] = acc[mf][nf][v] + bias[n];
      }
}

// ---------------------------------------------------------------------------
// Fused flash attention, fully software-pipelined (single barrier / tile).
// Q pre-scaled by 1/sqrt(d) (folded into Wq).  Per block: 128 query rows of
// one (b,h); 8 waves, wave = 16 rows x d64.  Key tile = 64, double-buffered:
// next tile's K goes async->LDS and V global->VGPR while computing current;
// V scatter into LDS happens after compute.  Q fragments hoisted (invariant).
// ---------------------------------------------------------------------------
__global__ __launch_bounds__(256)
void attn_wmma(const half_t* __restrict__ Q, const half_t* __restrict__ K,
               const half_t* __restrict__ V, half_t* __restrict__ O,
               int NQ, int NK) {
  constexpr int H = 8, DH = 64;
  constexpr int QP = 72, KP = 72, VP = 72, PP = 72;
  __shared__ __align__(16) half_t Qs[128 * QP];
  __shared__ __align__(16) half_t Ks[2][64 * KP];
  __shared__ __align__(16) half_t Vts[2][64 * VP];   // V^T tile: d x key
  __shared__ __align__(16) half_t Ps[8 * 16 * PP];

  const int tid  = threadIdx.x;
  const int lane = tid & 31;
  const int w    = tid >> 5;
  const int bi   = blockIdx.y / H;
  const int h    = blockIdx.y % H;
  const int q0   = blockIdx.x * 128;
  const size_t qbase = (size_t)bi * NQ * 512 + (size_t)h * DH;
  const size_t kbase = (size_t)bi * NK * 512 + (size_t)h * DH;

  const int vr_row = tid >> 2;            // V staging: my key row
  const int vr_col = (tid & 3) * 16;      // my 16 d-columns

  auto stage_k = [&](int buf, int kb) {   // 512 x b128, 2 per thread, async
#pragma unroll
    for (int i = 0; i < 2; i++) {
      int j = tid * 2 + i;
      int r = j >> 3, s = (j & 7) * 8;
      async_b128(lds_off(&Ks[buf][r * KP + s]),
                 K + kbase + (size_t)(kb + r) * 512 + s);
    }
  };
  auto load_v = [&](int kb, v8h& v0, v8h& v1) {
    const half_t* gv = V + kbase + (size_t)(kb + vr_row) * 512 + vr_col;
    v0 = ((const v8h*)gv)[0];
    v1 = ((const v8h*)gv)[1];
  };
  auto store_v = [&](int buf, v8h v0, v8h v1) {  // transpose into LDS
#pragma unroll
    for (int i = 0; i < 8; i++) {
      Vts[buf][(vr_col + i) * VP + vr_row]     = v0[i];
      Vts[buf][(vr_col + 8 + i) * VP + vr_row] = v1[i];
    }
  };

  // ---- Prologue: Q async, tile 0 K async + V through VGPRs ----
#pragma unroll
  for (int i = 0; i < 4; i++) {
    int j = tid * 4 + i;
    int r = j >> 3, s = (j & 7) * 8;
    async_b128(lds_off(&Qs[r * QP + s]),
               Q + qbase + (size_t)(q0 + r) * 512 + s);
  }
  stage_k(0, 0);
  {
    v8h v0, v1;
    load_v(0, v0, v1);
    wait_async0();
    store_v(0, v0, v1);
  }
  __syncthreads();

  const int ksel = (lane < 16) ? 0 : 8;
  const int kgrp = (lane >> 4) * 8;

  // Hoist loop-invariant Q A-fragments (16 q-rows x 64 d = 2 frags)
  v16h qfr[2];
#pragma unroll
  for (int ks = 0; ks < 2; ks++) {
    int qr = w * 16 + (lane & 15);
    v8h lo = *((const v8h*)&Qs[qr * QP + ks * 32 + ksel]);
    v8h hi = *((const v8h*)&Qs[qr * QP + ks * 32 + ksel + 16]);
    qfr[ks] = cat8(lo, hi);
  }

  v8f oacc[4];
  v8f zero = {};
#pragma unroll
  for (int o = 0; o < 4; o++) oacc[o] = zero;
  float mrow[8], ssum[8];
#pragma unroll
  for (int v = 0; v < 8; v++) { mrow[v] = -3.0e38f; ssum[v] = 0.0f; }

  int buf = 0;
  for (int kb = 0; kb < NK; kb += 64, buf ^= 1) {
    const bool more = (kb + 64) < NK;
    v8h nv0, nv1;
    if (more) {               // prefetch next tile under current compute
      stage_k(buf ^ 1, kb + 64);
      load_v(kb + 64, nv0, nv1);
    }

    // S = Q K^T : 16 q-rows x 64 keys per wave (4 N-frags, 2 K-steps)
    v8f sfr[4];
#pragma unroll
    for (int nf = 0; nf < 4; nf++) sfr[nf] = zero;
#pragma unroll
    for (int ks = 0; ks < 2; ks++) {
#pragma unroll
      for (int nf = 0; nf < 4; nf++) {
        int kr = nf * 16 + (lane & 15);
        v8h blo = *((const v8h*)&Ks[buf][kr * KP + ks * 32 + kgrp]);
        v8h bhi = *((const v8h*)&Ks[buf][kr * KP + ks * 32 + kgrp + 16]);
        sfr[nf] = WMMA_F32_F16(qfr[ks], cat8(blo, bhi), sfr[nf]);
      }
    }

    // Online softmax (row = v + 8*(lane>>4); DPP16 reductions per half)
#pragma unroll
    for (int v = 0; v < 8; v++) {
      float a0 = sfr[0][v], a1 = sfr[1][v], a2 = sfr[2][v], a3 = sfr[3][v];
      float t = fmaxf(fmaxf(a0, a1), fmaxf(a2, a3));
      t = row_max16(t);
      float mnew  = fmaxf(mrow[v], t);
      float alpha = __expf(mrow[v] - mnew);
      mrow[v] = mnew;
      float p0 = __expf(a0 - mnew), p1 = __expf(a1 - mnew);
      float p2 = __expf(a2 - mnew), p3 = __expf(a3 - mnew);
      float rs = row_sum16((p0 + p1) + (p2 + p3));
      ssum[v] = ssum[v] * alpha + rs;
#pragma unroll
      for (int o = 0; o < 4; o++) oacc[o][v] *= alpha;
      // bounce P through per-wave LDS region to re-layout C->A fragment
      int r = (w * 16 + v + ((lane >> 4) << 3)) * PP + (lane & 15);
      Ps[r]      = (half_t)p0;
      Ps[r + 16] = (half_t)p1;
      Ps[r + 32] = (half_t)p2;
      Ps[r + 48] = (half_t)p3;
    }

    // O += P V  (two 32-key A-frags of P; B-frags from V^T)
    {
      int pr = w * 16 + (lane & 15);
#pragma unroll
      for (int kf = 0; kf < 64; kf += 32) {
        v8h plo = *((const v8h*)&Ps[pr * PP + kf + ksel]);
        v8h phi = *((const v8h*)&Ps[pr * PP + kf + ksel + 16]);
        v16h pfr = cat8(plo, phi);
#pragma unroll
        for (int o = 0; o < 4; o++) {
          int vr = o * 16 + (lane & 15);
          v8h blo = *((const v8h*)&Vts[buf][vr * VP + kf + kgrp]);
          v8h bhi = *((const v8h*)&Vts[buf][vr * VP + kf + kgrp + 16]);
          oacc[o] = WMMA_F32_F16(pfr, cat8(blo, bhi), oacc[o]);
        }
      }
    }

    if (more) store_v(buf ^ 1, nv0, nv1);  // V scatter after compute
    wait_async0();
    __syncthreads();
  }

  // Epilogue: divide by row sums, store f16
#pragma unroll
  for (int v = 0; v < 8; v++) {
    float inv = 1.0f / ssum[v];
    int r = v + ((lane >> 4) << 3);
    size_t row = qbase + (size_t)(q0 + w * 16 + r) * 512;
#pragma unroll
    for (int o = 0; o < 4; o++)
      O[row + o * 16 + (lane & 15)] = (half_t)(oacc[o][v] * inv);
  }
}

// ---------------------------------------------------------------------------
// Host-side orchestration
// ---------------------------------------------------------------------------
extern "C" void kernel_launch(void* const* d_in, const int* in_sizes, int n_in,
                              void* d_out, int out_size, void* d_ws, size_t ws_size,
                              hipStream_t stream) {
  const float* pixel = (const float*)d_in[0];
  const float* patch = (const float*)d_in[1];
  const float* Wq    = (const float*)d_in[2];
  const float* Wk    = (const float*)d_in[3];
  const float* Wv    = (const float*)d_in[4];
  const float* Wo    = (const float*)d_in[5];
  const float* bo    = (const float*)d_in[6];
  float* out = (float*)d_out;

  const int Bc = 4, NQ = 2048, NK = 2048, DM = 512;
  const size_t actE = (size_t)Bc * NQ * DM;
  const size_t wE   = (size_t)DM * DM;

  char* ws = (char*)d_ws;
  size_t off = 0;
  auto alloc = [&](size_t bytes) -> half_t* {
    half_t* p = (half_t*)(ws + off);
    off += (bytes + 255) & ~(size_t)255;
    return p;
  };
  half_t* Xp  = alloc(actE * 2);
  half_t* Xc  = alloc(actE * 2);
  half_t* WqT = alloc(wE * 2);
  half_t* WkT = alloc(wE * 2);
  half_t* WvT = alloc(wE * 2);
  half_t* WoT = alloc(wE * 2);
  half_t* Q16 = alloc(actE * 2);
  half_t* K16 = alloc(actE * 2);
  half_t* V16 = alloc(actE * 2);
  half_t* AO  = Xp;  // pixel f16 dead after Q projection

  {
    int n4 = (int)(actE / 4);
    int blocks = (n4 + 255) / 256;
    cvt_f32_to_f16<<<blocks, 256, 0, stream>>>(pixel, Xp, n4);
    cvt_f32_to_f16<<<blocks, 256, 0, stream>>>(patch, Xc, n4);
  }
  {
    dim3 g(DM / 256, DM);
    // fold 1/sqrt(64) into Wq so attention needs no score scaling
    cvt_transpose_f16<<<g, 256, 0, stream>>>(Wq, WqT, DM, DM, 0.125f);
    cvt_transpose_f16<<<g, 256, 0, stream>>>(Wk, WkT, DM, DM, 1.0f);
    cvt_transpose_f16<<<g, 256, 0, stream>>>(Wv, WvT, DM, DM, 1.0f);
    cvt_transpose_f16<<<g, 256, 0, stream>>>(Wo, WoT, DM, DM, 1.0f);
  }
  {
    dim3 g(Bc * NQ / 128, DM / 128);
    gemm_wmma<1><<<g, 256, 0, stream>>>(Xp, WqT, Q16, nullptr, nullptr, Bc * NQ, DM, DM);
    gemm_wmma<1><<<g, 256, 0, stream>>>(Xc, WkT, K16, nullptr, nullptr, Bc * NK, DM, DM);
    gemm_wmma<1><<<g, 256, 0, stream>>>(Xc, WvT, V16, nullptr, nullptr, Bc * NK, DM, DM);
  }
  {
    dim3 g(NQ / 128, Bc * 8);
    attn_wmma<<<g, 256, 0, stream>>>(Q16, K16, V16, AO, NQ, NK);
  }
  {
    dim3 g(Bc * NQ / 128, DM / 128);
    gemm_wmma<0><<<g, 256, 0, stream>>>(AO, WoT, nullptr, out, bo, Bc * NQ, DM, DM);
  }
}
